// LocalAttentionBlock_43963285242534
// MI455X (gfx1250) — matrix operand order repaired and checked by reading, not compiled, not run on hardware
//
#include <hip/hip_runtime.h>

// ---------------------------------------------------------------------------
// Local attention block for MI455X (gfx1250), wave32, bf16 WMMA pipeline.
// Async global->LDS staging (ASYNCcnt), double buffering, all WMMA operands
// K-contiguous (b128 LDS loads, no u16 gathers).
// ---------------------------------------------------------------------------

#define B_SZ     2
#define L_SEQ    2048
#define D_MODEL  2048
#define N_HEADS  16
#define HD       128
#define WIN      512
#define M_ROWS   (B_SZ * L_SEQ)   // 4096
#define NUMQB    (L_SEQ / 64)     // 32

typedef __attribute__((ext_vector_type(16))) __bf16 v16bf;
typedef __attribute__((ext_vector_type(8)))  float  v8f;
typedef __attribute__((ext_vector_type(4)))  int    v4i_t;

// ---------------------------------------------------------------------------
// CDNA5 async global->LDS copy (GLOBAL_LOAD_ASYNC_TO_LDS_B128, ASYNCcnt)
// ---------------------------------------------------------------------------
typedef __attribute__((address_space(1))) v4i_t* as1_v4i_p;
typedef __attribute__((address_space(3))) v4i_t* as3_v4i_p;

// low 32 bits of a generic pointer into LDS == LDS byte offset (ISA 10.2)
__device__ __forceinline__ unsigned lds_off(const void* p) {
  return (unsigned)(uintptr_t)p;
}

__device__ __forceinline__ void async_b128(unsigned lds, const void* g) {
#if __has_builtin(__builtin_amdgcn_global_load_async_to_lds_b128)
  __builtin_amdgcn_global_load_async_to_lds_b128(
      (as1_v4i_p)(uintptr_t)g, (as3_v4i_p)(uintptr_t)lds, 0, 0);
#else
  asm volatile("global_load_async_to_lds_b128 %0, %1, off"
               :: "v"(lds), "v"((unsigned long long)(uintptr_t)g)
               : "memory");
#endif
}

#if __has_builtin(__builtin_amdgcn_s_wait_asynccnt)
#define WAIT_ASYNC(n) __builtin_amdgcn_s_wait_asynccnt(n)
#else
#define WAIT_ASYNC(n) asm volatile("s_wait_asynccnt %0" :: "i"(n) : "memory")
#endif

// ---------------------------------------------------------------------------
// WMMA fragment loaders (wave32; A and B both use contiguous k = 16*(lane/16) + s).
// Both loaders read K-contiguous pairs -> vectorize to ds_load_b128.
// ---------------------------------------------------------------------------

// A fragment: 16x32 (MxK), row-major source (M x K) with stride ld.
__device__ __forceinline__ v16bf load_frag_a(const __bf16* p, int ld) {
  int lane = threadIdx.x & 31;
  int m  = lane & 15;
  int kb = (lane >> 4) << 4;
  union { v16bf v; unsigned u[8]; } f;
#pragma unroll
  for (int r = 0; r < 8; ++r) {
    int k = kb + 2 * r;
    f.u[r] = *(const unsigned*)&p[m * ld + k];
  }
  return f.v;
}

// B fragment: 32x16 (KxN) from an N-MAJOR source: element (k,n) = p[n*ld + k].
__device__ __forceinline__ v16bf load_frag_b_t(const __bf16* p, int ld) {
  int lane = threadIdx.x & 31;
  int n  = lane & 15;
  int kb = (lane >> 4) << 4;
  union { v16bf v; unsigned u[8]; } f;
#pragma unroll
  for (int v = 0; v < 8; ++v)
    f.u[v] = *(const unsigned*)&p[n * ld + kb + 2 * v];
  return f.v;
}

__device__ __forceinline__ v8f wmma_bf16(v16bf a, v16bf b, v8f c) {
  return __builtin_amdgcn_wmma_f32_16x16x32_bf16(false, a, false, b,
                                                 (short)0, c, false, false);
}

// ---------------------------------------------------------------------------
// f32 -> bf16 conversion (row-major copy)
// ---------------------------------------------------------------------------
__global__ void lab_cvt_bf16(const float* __restrict__ src,
                             __bf16* __restrict__ dst, int n) {
  int i = blockIdx.x * blockDim.x + threadIdx.x;
  if (i < n) dst[i] = (__bf16)src[i];
}

// f32 [K x N] -> bf16 transposed [N x K] (weights become N-major)
__global__ void lab_cvt_bf16_t(const float* __restrict__ W,
                               __bf16* __restrict__ Wt, int K, int N) {
  int idx = blockIdx.x * blockDim.x + threadIdx.x;
  if (idx >= K * N) return;
  int n = idx / K;
  int k = idx - n * K;
  Wt[idx] = (__bf16)W[(size_t)k * N + n];
}

// f32 V [B*L x HD] -> bf16 Vt [B][HD][L] (dim-major per batch)
__global__ void lab_cvt_v_t(const float* __restrict__ V,
                            __bf16* __restrict__ Vt, int total) {
  int idx = blockIdx.x * blockDim.x + threadIdx.x;
  if (idx >= total) return;
  int l = idx & (L_SEQ - 1);
  int d = (idx / L_SEQ) & (HD - 1);
  int b = idx / (L_SEQ * HD);
  Vt[idx] = (__bf16)V[((size_t)(b * L_SEQ + l)) * HD + d];
}

// ---------------------------------------------------------------------------
// RoPE (dims 0..63 of each head) + convert to bf16.
// ---------------------------------------------------------------------------
__global__ void lab_rope_cvt(const float* __restrict__ src,
                             __bf16* __restrict__ dst,
                             int total, int cols) {
  int idx = blockIdx.x * blockDim.x + threadIdx.x;
  if (idx >= total) return;
  int row = idx / cols;
  int c   = idx - row * cols;
  int d   = c & (HD - 1);
  float pos = (float)(row & (L_SEQ - 1));
  const float nl = -9.2103403719761836f;  // -ln(10000)
  float val;
  if (d < 32) {
    float fr = __expf(nl * (float)(2 * d) * (1.0f / 64.0f));
    float th = pos * fr;
    float x1 = src[idx], x2 = src[idx + 32];
    val = x1 * __cosf(th) - x2 * __sinf(th);
  } else if (d < 64) {
    int i2 = d - 32;
    float fr = __expf(nl * (float)(2 * i2) * (1.0f / 64.0f));
    float th = pos * fr;
    float x1 = src[idx - 32], x2 = src[idx];
    val = x1 * __sinf(th) + x2 * __cosf(th);
  } else {
    val = src[idx];
  }
  dst[idx] = (__bf16)val;
}

// ---------------------------------------------------------------------------
// bf16 GEMM: C[MxN] = A[MxK] @ Bt^T (+bias), f32 out.  Bt is N-major [N x K].
// 256 threads (8 waves, 4x2), block tile 128x128, K-step 32.
// Async double-buffered LDS staging; 8 WMMA per wave per stage.
// ---------------------------------------------------------------------------
__global__ void lab_gemm_bf16(const __bf16* __restrict__ A,
                              const __bf16* __restrict__ Bt,
                              float* __restrict__ C,
                              const float* __restrict__ bias,
                              int M, int N, int K) {
  __shared__ __align__(16) __bf16 As[2][128 * 32];   // 2 x 8 KB
  __shared__ __align__(16) __bf16 Bs[2][128 * 32];   // 2 x 8 KB (n-major)
  int tid  = threadIdx.x;
  int wave = tid >> 5;
  int lane = tid & 31;
  int nlo  = lane & 15;
  int hi   = lane >> 4;
  int m0 = blockIdx.y * 128;
  int n0 = blockIdx.x * 128;
  int wm = (wave & 3) * 32;     // 4 m-waves
  int wn = (wave >> 2) * 64;    // 2 n-waves

  // per-thread staging coordinates: 2 chunks each for A and Bt (4 async/stage)
  int r0 = tid >> 2,         c0 = (tid & 3) << 3;
  int r1 = (256 + tid) >> 2, c1 = ((256 + tid) & 3) << 3;

  auto stage = [&](int buf, int k0) {
    async_b128(lds_off(&As[buf][r0 * 32 + c0]),
               &A[(size_t)(m0 + r0) * K + k0 + c0]);
    async_b128(lds_off(&As[buf][r1 * 32 + c1]),
               &A[(size_t)(m0 + r1) * K + k0 + c1]);
    async_b128(lds_off(&Bs[buf][r0 * 32 + c0]),
               &Bt[(size_t)(n0 + r0) * K + k0 + c0]);
    async_b128(lds_off(&Bs[buf][r1 * 32 + c1]),
               &Bt[(size_t)(n0 + r1) * K + k0 + c1]);
  };

  v8f acc[2][4] = {};
  int nst = K >> 5;
  stage(0, 0);

  for (int it = 0; it < nst; ++it) {
    int buf = it & 1;
    if (it + 1 < nst) {
      stage(buf ^ 1, (it + 1) << 5);   // overlaps with compute below
      WAIT_ASYNC(4);                   // current buffer landed (in-order)
    } else {
      WAIT_ASYNC(0);
    }
    __syncthreads();

    v16bf a[2], bfr[4];
#pragma unroll
    for (int mi = 0; mi < 2; ++mi)
      a[mi] = load_frag_a(&As[buf][(wm + mi * 16) * 32], 32);
#pragma unroll
    for (int ni = 0; ni < 4; ++ni)
      bfr[ni] = load_frag_b_t(&Bs[buf][(wn + ni * 16) * 32], 32);
#pragma unroll
    for (int mi = 0; mi < 2; ++mi)
#pragma unroll
      for (int ni = 0; ni < 4; ++ni)
        acc[mi][ni] = wmma_bf16(a[mi], bfr[ni], acc[mi][ni]);

    __syncthreads();   // all waves done reading buf before it is restaged
  }

#pragma unroll
  for (int mi = 0; mi < 2; ++mi)
#pragma unroll
    for (int ni = 0; ni < 4; ++ni)
#pragma unroll
      for (int r = 0; r < 8; ++r) {
        int row = m0 + wm + mi * 16 + r + 8 * hi;
        int col = n0 + wn + ni * 16 + nlo;
        float v = acc[mi][ni][r];
        if (bias) v += bias[col];
        C[(size_t)row * N + col] = v;
      }
}

// ---------------------------------------------------------------------------
// Flash attention, 512-token causal window, MQA (1 KV head).
// Grid: B*H*(L/64) blocks of 128 threads; wave w owns 16 query rows.
// K (token-major) and Vt (dim-major) double-buffered via async global->LDS.
// ---------------------------------------------------------------------------
__global__ void lab_attn(const __bf16* __restrict__ Qbf,
                         const __bf16* __restrict__ Kbf,
                         const __bf16* __restrict__ Vt,
                         __bf16* __restrict__ Obf) {
  __shared__ __align__(16) __bf16 Qs[64 * HD];          // 16 KB
  __shared__ __align__(16) __bf16 Ks[2][32 * HD];       // 16 KB (token-major)
  __shared__ __align__(16) __bf16 Vs[2][HD * 32];       // 16 KB (dim-major)
  __shared__ __align__(16) __bf16 Ps[4][16 * 32];       //  4 KB

  int tid  = threadIdx.x;
  int wave = tid >> 5;
  int lane = tid & 31;
  int nlo  = lane & 15;
  int hi   = lane >> 4;

  int bid = blockIdx.x;
  int qb  = bid & (NUMQB - 1);
  int h   = (bid / NUMQB) & (N_HEADS - 1);
  int b   = bid / (NUMQB * N_HEADS);
  int q0  = qb * 64;

  const __bf16* Qbase  = Qbf + ((size_t)(b * L_SEQ + q0)) * D_MODEL + h * HD;
  const __bf16* Kbase  = Kbf + ((size_t)(b * L_SEQ)) * HD;
  const __bf16* Vtbase = Vt + (size_t)b * HD * L_SEQ;

  // stage 64x128 Q block (8 async b128 per thread)
#pragma unroll
  for (int i = 0; i < 8; ++i) {
    int c = i * 128 + tid;
    int row = c >> 4, col = (c & 15) << 3;
    async_b128(lds_off(&Qs[row * HD + col]),
               &Qbase[(size_t)row * D_MODEL + col]);
  }

  int kstart = max(0, q0 - (WIN - 1)) & ~31;
  int nkb    = ((q0 + 64) - kstart) >> 5;

  auto stageKV = [&](int buf, int kb) {
    const __bf16* Kb = Kbase + (size_t)kb * HD;
#pragma unroll
    for (int i = 0; i < 4; ++i) {
      int c = i * 128 + tid;
      int krow = c >> 4, kcol = (c & 15) << 3;    // K tile: 32 tok x 128 dim
      async_b128(lds_off(&Ks[buf][krow * HD + kcol]),
                 &Kb[(size_t)krow * HD + kcol]);
      int vrow = c >> 2, vcol = (c & 3) << 3;     // V tile: 128 dim x 32 tok
      async_b128(lds_off(&Vs[buf][vrow * 32 + vcol]),
                 &Vtbase[(size_t)vrow * L_SEQ + kb + vcol]);
    }
  };

  stageKV(0, kstart);
  WAIT_ASYNC(8);          // Q complete (async loads retire in order)
  __syncthreads();

  v16bf qf[4];
#pragma unroll
  for (int kk = 0; kk < 4; ++kk)
    qf[kk] = load_frag_a(&Qs[(wave * 16) * HD + kk * 32], HD);

  v8f o[8] = {};
  float mrow[8], lrow[8];
#pragma unroll
  for (int r = 0; r < 8; ++r) { mrow[r] = -1e30f; lrow[r] = 0.0f; }

  const float scale = 0.08838834764831843f;  // 1/sqrt(128)

  for (int it = 0; it < nkb; ++it) {
    int kb  = kstart + it * 32;
    int buf = it & 1;
    if (it + 1 < nkb) {
      stageKV(buf ^ 1, kb + 32);  // overlaps with this block's compute
      WAIT_ASYNC(8);
    } else {
      WAIT_ASYNC(0);
    }
    __syncthreads();

    // S(16x32) = Q(16x128) @ K^T(128x32)
    v8f s[2] = {};
#pragma unroll
    for (int kk = 0; kk < 4; ++kk) {
      v16bf bf0 = load_frag_b_t(&Ks[buf][0 * HD + kk * 32], HD);
      v16bf bf1 = load_frag_b_t(&Ks[buf][16 * HD + kk * 32], HD);
      s[0] = wmma_bf16(qf[kk], bf0, s[0]);
      s[1] = wmma_bf16(qf[kk], bf1, s[1]);
    }

    // online softmax update
#pragma unroll
    for (int r = 0; r < 8; ++r) {
      int qi = q0 + wave * 16 + r + 8 * hi;
      int j0 = kb + nlo, j1 = kb + 16 + nlo;
      int d0 = qi - j0, d1 = qi - j1;
      bool v0 = (d0 >= 0) && (d0 < WIN);
      bool v1 = (d1 >= 0) && (d1 < WIN);
      float s0 = v0 ? s[0][r] * scale : -1e9f;
      float s1 = v1 ? s[1][r] * scale : -1e9f;
      float mx = fmaxf(s0, s1);
#pragma unroll
      for (int off = 1; off < 16; off <<= 1)
        mx = fmaxf(mx, __shfl_xor(mx, off, 32));
      float mnew = fmaxf(mrow[r], mx);
      float p0 = v0 ? __expf(s0 - mnew) : 0.0f;  // masked lanes exactly 0
      float p1 = v1 ? __expf(s1 - mnew) : 0.0f;
      float ps = p0 + p1;
#pragma unroll
      for (int off = 1; off < 16; off <<= 1)
        ps += __shfl_xor(ps, off, 32);
      float corr = __expf(mrow[r] - mnew);
      lrow[r] = lrow[r] * corr + ps;
      mrow[r] = mnew;
#pragma unroll
      for (int ni = 0; ni < 8; ++ni) o[ni][r] *= corr;
      Ps[wave][(r + 8 * hi) * 32 + nlo]      = (__bf16)p0;
      Ps[wave][(r + 8 * hi) * 32 + 16 + nlo] = (__bf16)p1;
    }

    // O += P(16x32) @ V(32x128)   (V dim-major -> contiguous b128 frags)
    v16bf pf = load_frag_a(Ps[wave], 32);
#pragma unroll
    for (int ni = 0; ni < 8; ++ni) {
      v16bf vf = load_frag_b_t(&Vs[buf][(ni * 16) * 32], 32);
      o[ni] = wmma_bf16(pf, vf, o[ni]);
    }

    __syncthreads();  // all waves done with buf before it is restaged
  }

  // normalize + store (b, l, h*hd + d) so the final GEMM sees B*L x D
  __bf16* Ob = Obf + ((size_t)(b * L_SEQ + q0 + wave * 16)) * D_MODEL + h * HD;
#pragma unroll
  for (int r = 0; r < 8; ++r) {
    float inv = 1.0f / lrow[r];
#pragma unroll
    for (int ni = 0; ni < 8; ++ni) {
      int row = r + 8 * hi;
      int col = ni * 16 + nlo;
      Ob[(size_t)row * D_MODEL + col] = (__bf16)(o[ni][r] * inv);
    }
  }
}

// ---------------------------------------------------------------------------
// Host launcher
// ---------------------------------------------------------------------------
extern "C" void kernel_launch(void* const* d_in, const int* in_sizes, int n_in,
                              void* d_out, int out_size, void* d_ws,
                              size_t ws_size, hipStream_t stream) {
  (void)in_sizes; (void)n_in; (void)out_size; (void)ws_size;
  const float* x  = (const float*)d_in[0];
  const float* Wq = (const float*)d_in[1];
  const float* Wk = (const float*)d_in[2];
  const float* Wv = (const float*)d_in[3];
  const float* Wo = (const float*)d_in[4];
  const float* bo = (const float*)d_in[5];
  float* out = (float*)d_out;

  char* ws = (char*)d_ws;
  size_t off = 0;
  auto alloc = [&](size_t bytes) -> void* {
    void* p = ws + off;
    off = (off + bytes + 255) & ~(size_t)255;
    return p;
  };

  const int M = M_ROWS, D = D_MODEL, hd = HD;
  __bf16* x_bf  = (__bf16*)alloc((size_t)M * D * 2);
  __bf16* Wqt   = (__bf16*)alloc((size_t)D * D * 2);    // [N=D x K=D]
  __bf16* Wkt   = (__bf16*)alloc((size_t)hd * D * 2);   // [N=hd x K=D]
  __bf16* Wvt   = (__bf16*)alloc((size_t)hd * D * 2);
  __bf16* Wot   = (__bf16*)alloc((size_t)D * D * 2);
  float*  Qraw  = (float*)alloc((size_t)M * D * 4);
  float*  Kraw  = (float*)alloc((size_t)M * hd * 4);
  float*  Vraw  = (float*)alloc((size_t)M * hd * 4);
  __bf16* Qbf   = (__bf16*)alloc((size_t)M * D * 2);
  __bf16* Kbf   = (__bf16*)alloc((size_t)M * hd * 2);
  __bf16* Vt    = (__bf16*)alloc((size_t)M * hd * 2);   // [B][HD][L]
  __bf16* Obf   = (__bf16*)alloc((size_t)M * D * 2);

  const int T = 256;
  lab_cvt_bf16<<<(M * D + T - 1) / T, T, 0, stream>>>(x, x_bf, M * D);
  lab_cvt_bf16_t<<<(D * D + T - 1) / T, T, 0, stream>>>(Wq, Wqt, D, D);
  lab_cvt_bf16_t<<<(D * hd + T - 1) / T, T, 0, stream>>>(Wk, Wkt, D, hd);
  lab_cvt_bf16_t<<<(D * hd + T - 1) / T, T, 0, stream>>>(Wv, Wvt, D, hd);
  lab_cvt_bf16_t<<<(D * D + T - 1) / T, T, 0, stream>>>(Wo, Wot, D, D);

  dim3 blk(256);
  lab_gemm_bf16<<<dim3(D / 128, M / 128), blk, 0, stream>>>(x_bf, Wqt, Qraw,
                                                            nullptr, M, D, D);
  lab_gemm_bf16<<<dim3(hd / 128, M / 128), blk, 0, stream>>>(x_bf, Wkt, Kraw,
                                                             nullptr, M, hd, D);
  lab_gemm_bf16<<<dim3(hd / 128, M / 128), blk, 0, stream>>>(x_bf, Wvt, Vraw,
                                                             nullptr, M, hd, D);

  lab_rope_cvt<<<(M * D + T - 1) / T, T, 0, stream>>>(Qraw, Qbf, M * D, D);
  lab_rope_cvt<<<(M * hd + T - 1) / T, T, 0, stream>>>(Kraw, Kbf, M * hd, hd);
  lab_cvt_v_t<<<(M * hd + T - 1) / T, T, 0, stream>>>(Vraw, Vt, M * hd);

  lab_attn<<<B_SZ * N_HEADS * NUMQB, 128, 0, stream>>>(Qbf, Kbf, Vt, Obf);

  lab_gemm_bf16<<<dim3(D / 128, M / 128), blk, 0, stream>>>(Obf, Wot, out,
                                                            bo, M, D, D);
}
